// RouterGate_64415919505632
// MI455X (gfx1250) — matrix operand order repaired and checked by reading, not compiled, not run on hardware
//
#include <hip/hip_runtime.h>
#include <hip/hip_bf16.h>

// ---------------------------------------------------------------------------
// MoE router + routed experts for MI455X (gfx1250), bf16 WMMA w/ f32 accum.
// Weights pre-transposed to N x K bf16 so BOTH GEMM operands stream via
// GLOBAL_LOAD_ASYNC_TO_LDS_B128 (ASYNCcnt) into double-buffered LDS.
// ---------------------------------------------------------------------------

typedef __attribute__((ext_vector_type(16))) __bf16          v16bf;
typedef __attribute__((ext_vector_type(8)))  float           v8f;
typedef __attribute__((ext_vector_type(8)))  unsigned short  v8us;
typedef __attribute__((ext_vector_type(16))) unsigned short  v16us;
typedef __attribute__((ext_vector_type(4)))  int             v4i;

#define BM 128
#define BN 128
#define BK 32
#define LDT 48   // LDS row stride in ushorts (96B; keeps 16B-aligned chunks)

#if __has_builtin(__builtin_amdgcn_global_load_async_to_lds_b128)
#define HAVE_ASYNC_LDS 1
typedef __attribute__((address_space(1))) v4i as1_v4i;   // global int4
typedef __attribute__((address_space(3))) v4i as3_v4i;   // LDS int4
#else
#define HAVE_ASYNC_LDS 0
#endif

static __device__ __forceinline__ void wait_async_all() {
#if HAVE_ASYNC_LDS
#if __has_builtin(__builtin_amdgcn_s_wait_asynccnt)
    __builtin_amdgcn_s_wait_asynccnt(0);
#else
    asm volatile("s_wait_asynccnt 0" ::: "memory");
#endif
#endif
}

static __device__ __forceinline__ unsigned short f2bf(float f) {
    unsigned u = __float_as_uint(f);
    u += 0x7FFFu + ((u >> 16) & 1u);   // round to nearest even
    return (unsigned short)(u >> 16);
}
static __device__ __forceinline__ float bf2f(unsigned short h) {
    return __uint_as_float(((unsigned)h) << 16);
}

// --------------------------- f32 -> bf16 convert ---------------------------
__global__ void cvt_bf16_kernel(const float* __restrict__ in,
                                unsigned short* __restrict__ out, int n) {
    int i = blockIdx.x * blockDim.x + threadIdx.x;
    int stride = gridDim.x * blockDim.x;
    for (; i < n; i += stride) out[i] = f2bf(in[i]);
}

// ----------------- f32 -> bf16 convert + transpose (R x C -> C x R) --------
// LDS-tiled 32x32 transpose, coalesced reads and writes; blockIdx.z = matrix.
__global__ __launch_bounds__(256)
void cvt_tr_bf16_kernel(const float* __restrict__ in,
                        unsigned short* __restrict__ out, int R, int C)
{
    __shared__ float tile[32][33];
    const float*    inp  = in  + (size_t)blockIdx.z * R * C;
    unsigned short* outp = out + (size_t)blockIdx.z * R * C;
    int c0 = blockIdx.x * 32;
    int r0 = blockIdx.y * 32;
    int x  = threadIdx.x & 31;
    int y  = threadIdx.x >> 5;   // 0..7
#pragma unroll
    for (int j = 0; j < 4; ++j) {
        int r = y + 8 * j;
        tile[r][x] = inp[(size_t)(r0 + r) * C + c0 + x];
    }
    __syncthreads();
#pragma unroll
    for (int j = 0; j < 4; ++j) {
        int r = y + 8 * j;       // output row within tile (= input col)
        outp[(size_t)(c0 + r) * R + r0 + x] = f2bf(tile[x][r]);
    }
}

// ----------------------- bf16 WMMA GEMM (multi-mode) -----------------------
// C[M,N] = A[M,K] @ W[K,N] + bias, with W provided TRANSPOSED (WT is N x K).
//   OMODE 0 : store bf16, compact rows
//   OMODE 1 : SiLU then store bf16, compact rows
//   OMODE 2 : atomicAdd(out_f32[o_idx[m]*N+col], o_wgt[m]*val)
//   a_idx   != null : A rows gathered (A row = a_idx[m])
//   p_count != null : M limited by *p_count (early-exit tiles)
template <int OMODE>
__global__ __launch_bounds__(256)
void moe_gemm_bf16(const unsigned short* __restrict__ A,
                   const unsigned short* __restrict__ WT,
                   const float* __restrict__ bias,
                   int Mmax, int N, int K,
                   const int* __restrict__ p_count,
                   const int* __restrict__ a_idx,
                   unsigned short* __restrict__ out_bf,
                   float* __restrict__ out_f32,
                   const int* __restrict__ o_idx,
                   const float* __restrict__ o_wgt)
{
    __shared__ __align__(16) unsigned short As[2][BM * LDT];
    __shared__ __align__(16) unsigned short Bs[2][BN * LDT];   // [n][k]

    int Mv = Mmax;
    if (p_count) { int c = *p_count; Mv = c < Mmax ? c : Mmax; }
    int m0 = blockIdx.y * BM;
    if (m0 >= Mv) return;
    int n0 = blockIdx.x * BN;

    const int tid  = threadIdx.x;
    const int lane = tid & 31;
    const int wave = tid >> 5;     // 0..7
    const int wr   = wave >> 1;    // 0..3 : wave row (32 rows each)
    const int wc   = wave & 1;     // 0..1 : wave col (64 cols each)
    const int half = lane >> 4;
    const int l16  = lane & 15;

    // ---- tile staging assignment: thread -> (row, 16-half k-segment) ----
    const int srow = tid >> 1;            // 0..127
    const int sseg = (tid & 1) * 16;      // 0 or 16 (halves)
    const int soff = srow * LDT + sseg;   // LDS dest offset (halves)

    // A rows (clamped gather: invalid rows read row 0; their outputs are
    // never stored, and C row m depends only on A row m => no contamination).
    int gm  = m0 + srow;
    int gmc = gm < Mv ? gm : 0;
    long grow = a_idx ? (long)a_idx[gmc] : (long)gmc;
    const unsigned short* Arow  = A  + (size_t)grow * K + sseg;
    // B rows of WT (N x K): tile row = output column n0+srow.
    const unsigned short* WTrow = WT + (size_t)(n0 + srow) * K + sseg;

    auto stage = [&](int buf, int k0) {
#if HAVE_ASYNC_LDS
        __builtin_amdgcn_global_load_async_to_lds_b128(
            (as1_v4i*)(Arow + k0),      (as3_v4i*)(&As[buf][soff]),     0, 0);
        __builtin_amdgcn_global_load_async_to_lds_b128(
            (as1_v4i*)(Arow + k0 + 8),  (as3_v4i*)(&As[buf][soff + 8]), 0, 0);
        __builtin_amdgcn_global_load_async_to_lds_b128(
            (as1_v4i*)(WTrow + k0),     (as3_v4i*)(&Bs[buf][soff]),     0, 0);
        __builtin_amdgcn_global_load_async_to_lds_b128(
            (as1_v4i*)(WTrow + k0 + 8), (as3_v4i*)(&Bs[buf][soff + 8]), 0, 0);
#else
        {
            const v8us* pa = (const v8us*)(Arow + k0);
            v8us a0 = pa[0], a1 = pa[1];
            const v8us* pb = (const v8us*)(WTrow + k0);
            v8us b0 = pb[0], b1 = pb[1];
            v8us* sa = (v8us*)(&As[buf][soff]);
            sa[0] = a0; sa[1] = a1;
            v8us* sb = (v8us*)(&Bs[buf][soff]);
            sb[0] = b0; sb[1] = b1;
        }
#endif
    };

    v8f acc[2][4];
#pragma unroll
    for (int i = 0; i < 2; ++i)
#pragma unroll
        for (int j = 0; j < 4; ++j) acc[i][j] = {};

    stage(0, 0);
    int buf = 0;
    for (int k0 = 0; k0 < K; k0 += BK, buf ^= 1) {
        wait_async_all();       // publish own async LDS writes before barrier
        __syncthreads();        // all staging for `buf` now visible

        if (k0 + BK < K) stage(buf ^ 1, k0 + BK);   // overlap with compute

        // ---- load fragments (ISA 16-bit A 16x32 / B 32x16 layouts) ----
        v16bf af[2];
#pragma unroll
        for (int fm = 0; fm < 2; ++fm) {
            const unsigned short* ab = &As[buf][(wr * 32 + fm * 16 + l16) * LDT];
            v8us lo = *(const v8us*)(ab + 8 * half);        // K = 8*half ..
            v8us hi = *(const v8us*)(ab + 16 + 8 * half);   // K = 16+8*half ..
            v16us v = __builtin_shufflevector(lo, hi, 0,1,2,3,4,5,6,7,
                                                      8,9,10,11,12,13,14,15);
            af[fm] = __builtin_bit_cast(v16bf, v);
        }
        v16bf bfv[4];
#pragma unroll
        for (int fn = 0; fn < 4; ++fn) {
            const unsigned short* bb = &Bs[buf][(wc * 64 + fn * 16 + l16) * LDT + 8 * half];
            v8us lo = *(const v8us*)(bb);
            v8us hi = *(const v8us*)(bb + 16);
            v16us v = __builtin_shufflevector(lo, hi, 0,1,2,3,4,5,6,7,
                                                      8,9,10,11,12,13,14,15);
            bfv[fn] = __builtin_bit_cast(v16bf, v);
        }

        // ---- 8x v_wmma_f32_16x16x32_bf16 ----
#pragma unroll
        for (int fm = 0; fm < 2; ++fm)
#pragma unroll
            for (int fn = 0; fn < 4; ++fn)
                acc[fm][fn] = __builtin_amdgcn_wmma_f32_16x16x32_bf16(
                    false, af[fm], false, bfv[fn], (short)0, acc[fm][fn],
                    false, false);
    }

    // ---- epilogue ----
    float bcol[4];
#pragma unroll
    for (int fn = 0; fn < 4; ++fn)
        bcol[fn] = bias[n0 + wc * 64 + fn * 16 + l16];

#pragma unroll
    for (int fm = 0; fm < 2; ++fm) {
#pragma unroll
        for (int j = 0; j < 8; ++j) {
            int row = m0 + wr * 32 + fm * 16 + half * 8 + j;
            if (row >= Mv) continue;
            int   trow = 0;
            float wv   = 0.0f;
            if constexpr (OMODE == 2) { trow = o_idx[row]; wv = o_wgt[row]; }
#pragma unroll
            for (int fn = 0; fn < 4; ++fn) {
                int col = n0 + wc * 64 + fn * 16 + l16;
                float v = acc[fm][fn][j] + bcol[fn];
                if constexpr (OMODE == 1)
                    v = v * __builtin_amdgcn_rcpf(1.0f + __expf(-v));  // SiLU
                if constexpr (OMODE == 2)
                    atomicAdd(out_f32 + (size_t)trow * N + col, wv * v);
                else
                    out_bf[(size_t)row * N + col] = f2bf(v);
            }
        }
    }
}

// --------------------------- gate / router kernel --------------------------
// One wave per token: logits = att@Wg+bg, softmax -> gate_prob, top-2,
// renormalize, push (token, weight) onto per-expert lists, sum importance.
__global__ __launch_bounds__(256)
void gate_kernel(const unsigned short* __restrict__ attb,
                 const float* __restrict__ Wg, const float* __restrict__ bg,
                 float* __restrict__ gate_prob,
                 int* __restrict__ counts, int* __restrict__ idx,
                 float* __restrict__ wgt, float* __restrict__ importance,
                 int Btok, int Dd)
{
    __shared__ float simp[8];
    if (threadIdx.x < 8) simp[threadIdx.x] = 0.0f;
    __syncthreads();

    const int lane = threadIdx.x & 31;
    const int wave = threadIdx.x >> 5;
    const int t = blockIdx.x * 8 + wave;

    float acc[8];
#pragma unroll
    for (int e = 0; e < 8; ++e) acc[e] = 0.0f;

    const unsigned short* arow = attb + (size_t)t * Dd;
    for (int d = lane; d < Dd; d += 32) {
        float a = bf2f(arow[d]);
        const float* wr = Wg + (size_t)d * 8;
#pragma unroll
        for (int e = 0; e < 8; ++e) acc[e] += a * wr[e];
    }
#pragma unroll
    for (int off = 16; off > 0; off >>= 1)
#pragma unroll
        for (int e = 0; e < 8; ++e)
            acc[e] += __shfl_xor(acc[e], off, 32);

    if (lane == 0) {
        float p[8];
        float mx = -3.4e38f;
#pragma unroll
        for (int e = 0; e < 8; ++e) { p[e] = acc[e] + bg[e]; mx = p[e] > mx ? p[e] : mx; }
        float s = 0.0f;
#pragma unroll
        for (int e = 0; e < 8; ++e) { p[e] = __expf(p[e] - mx); s += p[e]; }
        float inv = 1.0f / s;
#pragma unroll
        for (int e = 0; e < 8; ++e) {
            p[e] *= inv;
            gate_prob[(size_t)t * 8 + e] = p[e];
            atomicAdd(&simp[e], p[e]);
        }
        // top-2 (ties -> lower index, like jax.lax.top_k)
        int i1 = 0; float p1 = p[0];
#pragma unroll
        for (int e = 1; e < 8; ++e) if (p[e] > p1) { p1 = p[e]; i1 = e; }
        int i2 = -1; float p2 = -3.4e38f;
#pragma unroll
        for (int e = 0; e < 8; ++e)
            if (e != i1 && p[e] > p2) { p2 = p[e]; i2 = e; }
        // renormalize via softmax over the two probabilities
        float e1 = __expf(p1), e2 = __expf(p2);
        float rs = 1.0f / (e1 + e2);
        float w1 = e1 * rs, w2 = e2 * rs;

        int s1 = atomicAdd(&counts[i1], 1);
        idx[(size_t)i1 * Btok + s1] = t;
        wgt[(size_t)i1 * Btok + s1] = w1;
        int s2 = atomicAdd(&counts[i2], 1);
        idx[(size_t)i2 * Btok + s2] = t;
        wgt[(size_t)i2 * Btok + s2] = w2;
    }
    __syncthreads();
    if (threadIdx.x < 8) atomicAdd(&importance[threadIdx.x], simp[threadIdx.x]);
}

// ------------------------------ loss kernel --------------------------------
__global__ void loss_kernel(const float* __restrict__ imp, float* __restrict__ out_loss)
{
    if (threadIdx.x == 0) {
        float s = 0.0f;
        for (int e = 0; e < 8; ++e) s += imp[e];
        float mean = s * 0.125f;
        float v = 0.0f;
        for (int e = 0; e < 8; ++e) { float d = imp[e] - mean; v += d * d; }
        v *= (1.0f / 7.0f);                 // ddof=1
        float r = sqrtf(v) / mean;
        *out_loss = 0.01f * r * r;
    }
}

// ------------------------------- launcher ----------------------------------
extern "C" void kernel_launch(void* const* d_in, const int* in_sizes, int n_in,
                              void* d_out, int out_size, void* d_ws, size_t ws_size,
                              hipStream_t stream)
{
    const int Btok = 16384, Dd = 1024, E = 8;

    const float* x  = (const float*)d_in[0];
    const float* q  = (const float*)d_in[1];
    const float* Wv = (const float*)d_in[2];
    const float* bv = (const float*)d_in[3];
    const float* Wo = (const float*)d_in[4];
    const float* bo = (const float*)d_in[5];
    const float* Wg = (const float*)d_in[6];
    const float* bg = (const float*)d_in[7];
    const float* W1 = (const float*)d_in[8];
    const float* b1 = (const float*)d_in[9];
    const float* W2 = (const float*)d_in[10];
    const float* b2 = (const float*)d_in[11];

    float* y         = (float*)d_out;
    float* gate_prob = y + (size_t)Btok * Dd;
    float* loss      = gate_prob + (size_t)Btok * E;

    char* w = (char*)d_ws;
    size_t off = 0;
    auto take = [&](size_t bytes) -> char* {
        char* p = w + off;
        off += (bytes + 255) & ~(size_t)255;
        return p;
    };
    unsigned short* qb   = (unsigned short*)take((size_t)Btok * Dd * 2); // aliased: attb
    unsigned short* xb   = (unsigned short*)take((size_t)Btok * Dd * 2);
    unsigned short* Vb   = (unsigned short*)take((size_t)Btok * Dd * 2); // aliased: hb
    unsigned short* W1t  = (unsigned short*)take((size_t)E * Dd * Dd * 2);
    unsigned short* W2t  = (unsigned short*)take((size_t)E * Dd * Dd * 2);
    unsigned short* Wvt  = (unsigned short*)take((size_t)Dd * Dd * 2);
    unsigned short* Wot  = (unsigned short*)take((size_t)Dd * Dd * 2);
    char*           ci   = take(256);
    int*   counts = (int*)ci;            // E ints
    float* imp    = (float*)(ci + 64);   // E floats
    int*   idxbuf = (int*)take((size_t)E * Btok * 4);
    float* wgtbuf = (float*)take((size_t)E * Btok * 4);

    // 1) bf16 conversions (activations plain; weights transposed to N x K)
    cvt_bf16_kernel<<<2048, 256, 0, stream>>>(q, qb, Btok * Dd);
    cvt_bf16_kernel<<<2048, 256, 0, stream>>>(x, xb, Btok * Dd);
    dim3 gt(Dd / 32, Dd / 32, 1);
    cvt_tr_bf16_kernel<<<gt, 256, 0, stream>>>(Wv, Wvt, Dd, Dd);
    cvt_tr_bf16_kernel<<<gt, 256, 0, stream>>>(Wo, Wot, Dd, Dd);
    dim3 gte(Dd / 32, Dd / 32, E);
    cvt_tr_bf16_kernel<<<gte, 256, 0, stream>>>(W1, W1t, Dd, Dd);
    cvt_tr_bf16_kernel<<<gte, 256, 0, stream>>>(W2, W2t, Dd, Dd);

    dim3 g(Dd / BN, Btok / BM);   // 8 x 128
    // 2) V = q@Wv + bv   (bf16 out)
    moe_gemm_bf16<0><<<g, 256, 0, stream>>>(qb, Wvt, bv, Btok, Dd, Dd,
                                            nullptr, nullptr, Vb,
                                            nullptr, nullptr, nullptr);
    // 3) att = V@Wo + bo (bf16 out, reuse qb)
    unsigned short* attb = qb;
    moe_gemm_bf16<0><<<g, 256, 0, stream>>>(Vb, Wot, bo, Btok, Dd, Dd,
                                            nullptr, nullptr, attb,
                                            nullptr, nullptr, nullptr);

    // 4) zero y, counts, importance
    (void)hipMemsetAsync(y, 0, (size_t)Btok * Dd * sizeof(float), stream);
    (void)hipMemsetAsync(ci, 0, 128, stream);

    // 5) gate + routing
    gate_kernel<<<Btok / 8, 256, 0, stream>>>(attb, Wg, bg, gate_prob,
                                              counts, idxbuf, wgtbuf, imp,
                                              Btok, Dd);

    // 6) routed experts: gather-GEMM1 (+SiLU) then GEMM2 with weighted scatter
    unsigned short* hb = Vb;
    for (int e = 0; e < E; ++e) {
        moe_gemm_bf16<1><<<g, 256, 0, stream>>>(
            xb, W1t + (size_t)e * Dd * Dd, b1 + (size_t)e * Dd,
            Btok, Dd, Dd, counts + e, idxbuf + (size_t)e * Btok,
            hb, nullptr, nullptr, nullptr);
        moe_gemm_bf16<2><<<g, 256, 0, stream>>>(
            hb, W2t + (size_t)e * Dd * Dd, b2 + (size_t)e * Dd,
            Btok, Dd, Dd, counts + e, nullptr,
            nullptr, y, idxbuf + (size_t)e * Btok, wgtbuf + (size_t)e * Btok);
    }

    // 7) importance loss
    loss_kernel<<<1, 32, 0, stream>>>(imp, loss);
}
